// EPN_34978213658963
// MI455X (gfx1250) — matrix operand order repaired and checked by reading, not compiled, not run on hardware
//
#include <hip/hip_runtime.h>
#include <hip/hip_bf16.h>

typedef __attribute__((ext_vector_type(16))) _Float16 v16h;
typedef __attribute__((ext_vector_type(8)))  _Float16 hv8;
typedef __attribute__((ext_vector_type(8)))  float    v8f;

#define LRELU_SLOPE 0.01f

__device__ __forceinline__ float leaky(float v) { return v > 0.0f ? v : LRELU_SLOPE * v; }

// ---------------------------------------------------------------------------
// Neighbor softmax weights + center extraction.
// xyz [4,3,Np], idx [4,M,16] -> w [4,M,16], xyzout [4,3,M]
// ---------------------------------------------------------------------------
__global__ void k_neighbor_w(const float* __restrict__ xyz, const int* __restrict__ idx,
                             float* __restrict__ wout, float* __restrict__ xyzout,
                             int Np, int M, float inv_sigma) {
    int m = blockIdx.x * blockDim.x + threadIdx.x;
    int b = blockIdx.y;
    if (m >= M) return;
    const int*   id = idx + (b * M + m) * 16;
    const float* xb = xyz + (size_t)b * 3 * Np;
    int   i0 = id[0];
    float cx = xb[i0], cy = xb[Np + i0], cz = xb[2 * Np + i0];
    float s[16];
    float mx = -1e30f;
#pragma unroll
    for (int k = 0; k < 16; ++k) {
        int i = id[k];
        float dx = xb[i] - cx, dy = xb[Np + i] - cy, dz = xb[2 * Np + i] - cz;
        float v = -(dx * dx + dy * dy + dz * dz) * inv_sigma;
        s[k] = v;
        mx = fmaxf(mx, v);
    }
    float sum = 0.0f;
#pragma unroll
    for (int k = 0; k < 16; ++k) { s[k] = __expf(s[k] - mx); sum += s[k]; }
    float inv = 1.0f / sum;
    float* wo = wout + (b * M + m) * 16;
#pragma unroll
    for (int k = 0; k < 16; ++k) wo[k] = s[k] * inv;
    float* xo = xyzout + (size_t)b * 3 * M;
    xo[m] = cx; xo[M + m] = cy; xo[2 * M + m] = cz;
}

// ---------------------------------------------------------------------------
// Layer 1a: Cin=1, feat==1  ->  out[b,a,m,o] = leaky(max_k W[o]*w[b,m,k])
// f16 output, channel-innermost layout [b][a][m][32].  grid (M/64, 4) block 64.
// ---------------------------------------------------------------------------
__global__ void k_conv1a(const float* __restrict__ W, const float* __restrict__ wbuf,
                         _Float16* __restrict__ out, int M) {
    int m = blockIdx.x * blockDim.x + threadIdx.x;
    int b = blockIdx.y;
    if (m >= M) return;
    const float* w = wbuf + (b * M + m) * 16;
    float mx = w[0], mn = w[0];
#pragma unroll
    for (int k = 1; k < 16; ++k) { mx = fmaxf(mx, w[k]); mn = fminf(mn, w[k]); }
    hv8 vals[4];
#pragma unroll
    for (int o = 0; o < 32; ++o) {
        float Wo = W[o];
        vals[o >> 3][o & 7] = (_Float16)leaky(Wo >= 0.0f ? Wo * mx : Wo * mn);
    }
    for (int a = 0; a < 60; ++a) {
        hv8* dst = (hv8*)(out + ((size_t)(b * 60 + a) * M + m) * 32);
        dst[0] = vals[0]; dst[1] = vals[1]; dst[2] = vals[2]; dst[3] = vals[3];
    }
}

// ---------------------------------------------------------------------------
// Pack f32 weight [Cout,Cin] into f16 WMMA A-fragment layout:
// pk[(t*nsteps+s)*512 + lane*16 + j]
//   lane<16 : row=t*16+lane,    col=s*32 + j       (K 0-15)
//   lane>=16: row=t*16+lane-16, col=s*32 + 16 + j  (K 16-31)
// ---------------------------------------------------------------------------
__global__ void k_pack(const float* __restrict__ W, _Float16* __restrict__ pk,
                       int Cin, int Cout) {
    int e = blockIdx.x * blockDim.x + threadIdx.x;
    if (e >= Cin * Cout) return;
    int nst = Cin >> 5;
    int j = e & 15;
    int L = (e >> 4) & 31;
    int s = (e >> 9) % nst;
    int t = e / (nst << 9);
    int row, col;
    if (L < 16) { row = t * 16 + L;        col = s * 32 + j;  }
    else        { row = t * 16 + (L - 16); col = s * 32 + 16 + j; }
    pk[e] = (_Float16)W[(size_t)row * Cin + col];
}

// ---------------------------------------------------------------------------
// WMMA point-conv layer. Features f16, layout [b][a][point][c] (c innermost):
// neighbor gather = contiguous-row copy -> pure 16B uint4 global->LDS staging.
// Softmax weight applied AFTER the GEMM (column k of the D-tile = neighbor k),
// then 16-lane cross-lane max = neighbor max-pool, leaky, 16B f16 store.
// grid: (M/NPTS, 60, 4), block 256 (8 waves; WPP waves per point).
// ---------------------------------------------------------------------------
template <int Cin, int Cout, int NPTS>
__global__ void k_conv_wmma(const _Float16* __restrict__ fin, const _Float16* __restrict__ pkW,
                            const int* __restrict__ idx, const float* __restrict__ wbuf,
                            _Float16* __restrict__ fout, int M, int Nprev) {
    constexpr int NSTEPS = Cin / 32;
    constexpr int NTILES = Cout / 16;
    constexpr int WPP    = 8 / NPTS;      // waves per point
    constexpr int PN     = NPTS * 16;
    constexpr int CH     = Cin / 8;       // 16-byte chunks per feature row
    __shared__ alignas(64) _Float16 gsh[NPTS * 16 * Cin];
    __shared__ int   ish[PN];
    __shared__ float wsh[PN];
    int m0 = blockIdx.x * NPTS;
    int a  = blockIdx.y;
    int b  = blockIdx.z;
    int tid = threadIdx.x;
    if (tid < PN) {
        int p = tid >> 4, n = tid & 15;
        ish[tid] = idx [(b * M + m0 + p) * 16 + n];
        wsh[tid] = wbuf[(b * M + m0 + p) * 16 + n];
    }
    __syncthreads();
    const _Float16* fbase = fin + (size_t)(b * 60 + a) * Nprev * Cin;
#pragma unroll 1
    for (int e = tid; e < PN * CH; e += 256) {
        int chunk = e & (CH - 1);          // compile-time mask
        int pn    = e / CH;                // compile-time shift
        *(uint4*)&gsh[pn * Cin + chunk * 8] =
            *(const uint4*)&fbase[(size_t)ish[pn] * Cin + chunk * 8];
    }
    __syncthreads();
    int lane = tid & 31, wave = tid >> 5;
    int p    = wave / WPP;                 // compile-time
    int sub  = wave % WPP;
    int ncol = lane & 15, hi = lane >> 4;
    const _Float16* gp = gsh + (p * 16 + ncol) * Cin + hi * 16;
    float wv = wsh[p * 16 + ncol];
    size_t orow = ((size_t)(b * 60 + a) * M + (m0 + p)) * Cout;
#pragma unroll 1
    for (int t = sub; t < NTILES; t += WPP) {
        v8f acc = {};
        const _Float16* ap = pkW + ((size_t)(t * NSTEPS) << 9) + (lane << 4);
#pragma unroll
        for (int s = 0; s < NSTEPS; ++s) {
            v16h af = *(const v16h*)(ap + (s << 9));
            v16h bf = *(const v16h*)(gp + (s << 5));
            acc = __builtin_amdgcn_wmma_f32_16x16x32_f16(false, af, false, bf,
                                                         (short)0, acc, false, false);
        }
        hv8 res;
#pragma unroll
        for (int r = 0; r < 8; ++r) {
            float v = acc[r] * wv;
            v = fmaxf(v, __shfl_xor(v, 1));
            v = fmaxf(v, __shfl_xor(v, 2));
            v = fmaxf(v, __shfl_xor(v, 4));
            v = fmaxf(v, __shfl_xor(v, 8));
            res[r] = (_Float16)leaky(v);
        }
        if (ncol == 0) *(hv8*)&fout[orow + t * 16 + hi * 8] = res;
    }
}

// ---------------------------------------------------------------------------
// WMMA channel MLP over 3840 flattened (a,m) columns. c-innermost f16 layout
// means each lane's B-fragment is 16 contiguous halves in global memory:
// no LDS, no barrier — pure VMEM -> WMMA.  grid (240, 4), block 256.
// ---------------------------------------------------------------------------
template <int Cin, int Cout>
__global__ void k_mlp_wmma(const _Float16* __restrict__ fin, const _Float16* __restrict__ pkW,
                           _Float16* __restrict__ fout) {
    constexpr int NSTEPS = Cin / 32;
    constexpr int NTILES = Cout / 16;
    int col0 = blockIdx.x * 16;
    int b    = blockIdx.y;
    int tid  = threadIdx.x;
    int lane = tid & 31, wave = tid >> 5;
    int ncol = lane & 15, hi = lane >> 4;
    const _Float16* gp = fin + ((size_t)b * 3840 + col0 + ncol) * Cin + hi * 16;
    size_t orow = ((size_t)b * 3840 + col0 + ncol) * Cout;
#pragma unroll 1
    for (int t = wave; t < NTILES; t += 8) {
        v8f acc = {};
        const _Float16* ap = pkW + ((size_t)(t * NSTEPS) << 9) + (lane << 4);
#pragma unroll
        for (int s = 0; s < NSTEPS; ++s) {
            v16h af = *(const v16h*)(ap + (s << 9));
            v16h bf = *(const v16h*)(gp + (s << 5));
            acc = __builtin_amdgcn_wmma_f32_16x16x32_f16(false, af, false, bf,
                                                         (short)0, acc, false, false);
        }
        hv8 res;
#pragma unroll
        for (int r = 0; r < 8; ++r) res[r] = (_Float16)leaky(acc[r]);
        *(hv8*)&fout[orow + t * 16 + hi * 8] = res;
    }
}

// ganc[b,c,a] = max_m g[b, col=a*64+m, c]  (f16 in [b][col][128] -> f32 out)
// grid (2, 60, 4), block 64
__global__ void k_ganc(const _Float16* __restrict__ g, float* __restrict__ ganc) {
    int c = blockIdx.x * 64 + threadIdx.x;
    int a = blockIdx.y;
    int b = blockIdx.z;
    const _Float16* p = g + ((size_t)(b * 3840 + a * 64)) * 128 + c;
    float mx = (float)p[0];
    for (int m = 1; m < 64; ++m) mx = fmaxf(mx, (float)p[(size_t)m * 128]);
    ganc[(b * 128 + c) * 60 + a] = mx;
}

// embed[b,c] = max_a ganc[b,c,a]
__global__ void k_embed(const float* __restrict__ ganc, float* __restrict__ embed) {
    int tid = blockIdx.x * blockDim.x + threadIdx.x;
    if (tid >= 512) return;
    const float* p = ganc + tid * 60;
    float mx = p[0];
    for (int i = 1; i < 60; ++i) mx = fmaxf(mx, p[i]);
    embed[tid] = mx;
}

// h[b,o,a] = leaky( sum_c Wa[o,c] * ganc[b,c,a] );  grid (1, 64, 4) block 64
__global__ void k_head_h(const float* __restrict__ Wa, const float* __restrict__ ganc,
                         float* __restrict__ h) {
    int a = blockIdx.x * blockDim.x + threadIdx.x;
    int o = blockIdx.y;
    int b = blockIdx.z;
    if (a >= 60) return;
    float s = 0.0f;
    for (int c = 0; c < 128; ++c) s += Wa[o * 128 + c] * ganc[(b * 128 + c) * 60 + a];
    h[(b * 64 + o) * 60 + a] = leaky(s);
}

// attention + quaternion + translation head; one block per batch
__global__ void k_pose(const float* __restrict__ Wattn, const float* __restrict__ Wq,
                       const float* __restrict__ Wt, const float* __restrict__ h,
                       float* __restrict__ quat_out, float* __restrict__ t_out) {
    __shared__ float attn[60];
    __shared__ float qs[4][60];
    __shared__ float ts[3][60];
    int b = blockIdx.x;
    int tid = threadIdx.x;
    const float* hb = h + b * 64 * 60;
    if (tid < 60) {
        float s = 0, q0 = 0, q1 = 0, q2 = 0, q3 = 0, t0 = 0, t1 = 0, t2 = 0;
        for (int c = 0; c < 64; ++c) {
            float hv = hb[c * 60 + tid];
            s  += Wattn[c] * hv;
            q0 += Wq[c] * hv;       q1 += Wq[64 + c] * hv;
            q2 += Wq[128 + c] * hv; q3 += Wq[192 + c] * hv;
            t0 += Wt[c] * hv; t1 += Wt[64 + c] * hv; t2 += Wt[128 + c] * hv;
        }
        attn[tid] = s / 3.0f;
        float nrm = sqrtf(q0 * q0 + q1 * q1 + q2 * q2 + q3 * q3) + 1e-8f;
        qs[0][tid] = q0 / nrm; qs[1][tid] = q1 / nrm;
        qs[2][tid] = q2 / nrm; qs[3][tid] = q3 / nrm;
        ts[0][tid] = t0; ts[1][tid] = t1; ts[2][tid] = t2;
    }
    __syncthreads();
    if (tid == 0) {
        float mx = attn[0];
        for (int a = 1; a < 60; ++a) mx = fmaxf(mx, attn[a]);
        float sum = 0.0f;
        for (int a = 0; a < 60; ++a) { attn[a] = __expf(attn[a] - mx); sum += attn[a]; }
        float inv = 1.0f / sum;
        float Q[4] = {0, 0, 0, 0}, T[3] = {0, 0, 0};
        for (int a = 0; a < 60; ++a) {
            float w = attn[a] * inv;
            Q[0] += qs[0][a] * w; Q[1] += qs[1][a] * w;
            Q[2] += qs[2][a] * w; Q[3] += qs[3][a] * w;
            T[0] += ts[0][a] * w; T[1] += ts[1][a] * w; T[2] += ts[2][a] * w;
        }
        float qn = sqrtf(Q[0]*Q[0] + Q[1]*Q[1] + Q[2]*Q[2] + Q[3]*Q[3]) + 1e-8f;
        for (int i = 0; i < 4; ++i) quat_out[b * 4 + i] = Q[i] / qn;
        for (int i = 0; i < 3; ++i) t_out[b * 3 + i] = T[i];
    }
}

// dense: out[b,o] = act( dot(in[b,:], W[o,:]) + bias[o] );  act: 0=leaky, 1=sigmoid
// grid (Cout/64, 4) block 64
__global__ void k_dense(const float* __restrict__ in, const float* __restrict__ W,
                        const float* __restrict__ bias, float* __restrict__ out,
                        int Cin, int Cout, int act) {
    int o = blockIdx.x * blockDim.x + threadIdx.x;
    int b = blockIdx.y;
    if (o >= Cout) return;
    float s = bias[o];
    const float* ib = in + b * Cin;
    const float* wr = W + (size_t)o * Cin;
    for (int c = 0; c < Cin; ++c) s += ib[c] * wr[c];
    out[b * Cout + o] = act ? 1.0f / (1.0f + __expf(-s)) : leaky(s);
}

// ---------------------------------------------------------------------------
extern "C" void kernel_launch(void* const* d_in, const int* in_sizes, int n_in,
                              void* d_out, int out_size, void* d_ws, size_t ws_size,
                              hipStream_t stream) {
    (void)in_sizes; (void)n_in; (void)out_size; (void)ws_size;
    const float* x     = (const float*)d_in[0];
    const int* idxA[8] = { (const int*)d_in[1], (const int*)d_in[2], (const int*)d_in[3],
                           (const int*)d_in[4], (const int*)d_in[5], (const int*)d_in[6],
                           (const int*)d_in[7], (const int*)d_in[8] };
    const float* W1a = (const float*)d_in[9];
    const float* Wc[7] = { (const float*)d_in[10], (const float*)d_in[11], (const float*)d_in[12],
                           (const float*)d_in[13], (const float*)d_in[14], (const float*)d_in[15],
                           (const float*)d_in[16] };
    const float* Wm1 = (const float*)d_in[17];
    const float* Wm2 = (const float*)d_in[18];
    const float* Wa  = (const float*)d_in[19];
    const float* Wattn = (const float*)d_in[20];
    const float* Wq  = (const float*)d_in[21];
    const float* Wt  = (const float*)d_in[22];
    const float* D1  = (const float*)d_in[23]; const float* b1 = (const float*)d_in[24];
    const float* D2  = (const float*)d_in[25]; const float* b2 = (const float*)d_in[26];
    const float* D3  = (const float*)d_in[27]; const float* b3 = (const float*)d_in[28];

    char* base = (char*)d_ws;
    _Float16* F0 = (_Float16*)base;  base += (size_t)4194304 * 2;   // f16 activations
    _Float16* F1 = (_Float16*)base;  base += (size_t)4194304 * 2;
    float* wbuf = (float*)base;  base += 32768 * 4;
    float* xyzA = (float*)base;  base += 8192 * 4;
    float* xyzB = (float*)base;  base += 8192 * 4;
    float* ganc = (float*)base;  base += 32768 * 4;
    float* hbuf = (float*)base;  base += 16384 * 4;
    float* dbf1 = (float*)base;  base += 1024 * 4;
    float* dbf2 = (float*)base;  base += 1024 * 4;
    _Float16* pkb = (_Float16*)base;
    _Float16* pk[9];
    size_t pkoff[9] = { 0, 1024, 3072, 7168, 15360, 31744, 64512, 130048, 162816 };
    for (int i = 0; i < 9; ++i) pk[i] = pkb + pkoff[i];

    // pack all weights (conv layers 1b..4b, Wm1, Wm2)
    const int pcin [9] = { 32, 32, 64, 64, 128, 128, 256, 256, 128 };
    const int pcout[9] = { 32, 64, 64, 128, 128, 256, 256, 128, 128 };
    const float* psrc[9] = { Wc[0], Wc[1], Wc[2], Wc[3], Wc[4], Wc[5], Wc[6], Wm1, Wm2 };
    for (int i = 0; i < 9; ++i) {
        int tot = pcin[i] * pcout[i];
        k_pack<<<(tot + 255) / 256, 256, 0, stream>>>(psrc[i], pk[i], pcin[i], pcout[i]);
    }

    // layer 1a (special: Cin=1, feat==1)
    k_neighbor_w<<<dim3(8, 4), 64, 0, stream>>>(x, idxA[0], wbuf, xyzA, 1024, 512,
                                                1.0f / 0.0032f);
    k_conv1a<<<dim3(8, 4), 64, 0, stream>>>(W1a, wbuf, F0, 512);

    // WMMA conv layers 1b..4b (ping-pong F0/F1); NPTS=8 (Cin<=128) else 4
    const int   lM [7] = { 512, 256, 256, 128, 128, 64, 64 };
    const int   lNp[7] = { 512, 512, 256, 256, 128, 128, 64 };
    const float lSg[7] = { 0.0064f, 0.0128f, 0.0128f, 0.0256f, 0.0256f, 0.0512f, 0.0512f };
    float* xin  = xyzA;
    float* xout = xyzB;
    _Float16* fin  = F0;
    _Float16* fout = F1;
    for (int l = 0; l < 7; ++l) {
        int M = lM[l], Np = lNp[l];
        k_neighbor_w<<<dim3((M + 63) / 64, 4), 64, 0, stream>>>(xin, idxA[l + 1], wbuf, xout,
                                                                Np, M, 1.0f / lSg[l]);
        dim3 grid8(M / 8, 60, 4), grid4(M / 4, 60, 4);
        switch (l) {
        case 0: k_conv_wmma< 32,  32, 8><<<grid8, 256, 0, stream>>>(fin, pk[0], idxA[1], wbuf, fout, M, Np); break;
        case 1: k_conv_wmma< 32,  64, 8><<<grid8, 256, 0, stream>>>(fin, pk[1], idxA[2], wbuf, fout, M, Np); break;
        case 2: k_conv_wmma< 64,  64, 8><<<grid8, 256, 0, stream>>>(fin, pk[2], idxA[3], wbuf, fout, M, Np); break;
        case 3: k_conv_wmma< 64, 128, 8><<<grid8, 256, 0, stream>>>(fin, pk[3], idxA[4], wbuf, fout, M, Np); break;
        case 4: k_conv_wmma<128, 128, 8><<<grid8, 256, 0, stream>>>(fin, pk[4], idxA[5], wbuf, fout, M, Np); break;
        case 5: k_conv_wmma<128, 256, 8><<<grid8, 256, 0, stream>>>(fin, pk[5], idxA[6], wbuf, fout, M, Np); break;
        case 6: k_conv_wmma<256, 256, 4><<<grid4, 256, 0, stream>>>(fin, pk[6], idxA[7], wbuf, fout, M, Np); break;
        }
        float* tx = xin; xin = xout; xout = tx;
        _Float16* tf = fin; fin = fout; fout = tf;
    }
    // after loop: final conv features are in `fin` (= F1), layout [b][a*64+m][256]

    // channel MLPs (Wm1: 256->128, Wm2: 128->128) over 3840 cols, LDS-free
    k_mlp_wmma<256, 128><<<dim3(240, 4), 256, 0, stream>>>(fin,  pk[7], fout);  // -> F0
    k_mlp_wmma<128, 128><<<dim3(240, 4), 256, 0, stream>>>(fout, pk[8], fin);   // -> F1

    // reductions + heads
    float* out_f = (float*)d_out;                 // [embed 512 | recon 12288 | quat 16 | t 12]
    k_ganc <<<dim3(2, 60, 4), 64, 0, stream>>>(fin, ganc);
    k_embed<<<2, 256, 0, stream>>>(ganc, out_f);
    k_head_h<<<dim3(1, 64, 4), 64, 0, stream>>>(Wa, ganc, hbuf);
    k_pose<<<4, 64, 0, stream>>>(Wattn, Wq, Wt, hbuf, out_f + 12800, out_f + 12816);

    // decoder
    k_dense<<<dim3(4, 4), 64, 0, stream>>>(out_f, D1, b1, dbf1, 128, 256, 0);
    k_dense<<<dim3(4, 4), 64, 0, stream>>>(dbf1, D2, b2, dbf2, 256, 256, 0);
    k_dense<<<dim3(48, 4), 64, 0, stream>>>(dbf2, D3, b3, out_f + 512, 256, 3072, 1);
}
